// GeneralAttention_58093727645748
// MI455X (gfx1250) — matrix operand order repaired and checked by reading, not compiled, not run on hardware
//
#include <hip/hip_runtime.h>

// Problem dimensions (fixed by the reference).
#define TQ 1024
#define TK 1024
#define BB 16
#define QD 1024
#define KD 1024
#define VD 1024

typedef __attribute__((ext_vector_type(16))) __bf16 v16bf;
typedef __attribute__((ext_vector_type(8)))  __bf16 v8bf;
typedef __attribute__((ext_vector_type(4)))  __bf16 v4bf;
typedef __attribute__((ext_vector_type(8)))  float  v8f;
typedef __attribute__((ext_vector_type(4)))  float  v4f;
typedef __attribute__((ext_vector_type(4)))  unsigned int v4u;
typedef __attribute__((ext_vector_type(8)))  int    v8i;
typedef __attribute__((ext_vector_type(4)))  int    v4i;

#if defined(__has_builtin)
#if __has_builtin(__builtin_amdgcn_tensor_load_to_lds) && \
    __has_builtin(__builtin_amdgcn_s_wait_tensorcnt)
#define HAVE_TDM 1
#endif
#endif

static __device__ __forceinline__ v8f wmma_bf16(v16bf a, v16bf b, v8f c) {
  return __builtin_amdgcn_wmma_f32_16x16x32_bf16(
      false, a, false, b, (short)0, c, false, false);
}

// ---------------------------------------------------------------------------
// f32 -> bf16 elementwise conversion (n is a multiple of 4).
// ---------------------------------------------------------------------------
__global__ __launch_bounds__(256) void cvt_f32_bf16(
    const float* __restrict__ in, __bf16* __restrict__ out, int n) {
  int i = (blockIdx.x * 256 + threadIdx.x) * 4;
  if (i + 3 < n) {
    v4f f = *(const v4f*)(in + i);
    v4bf h;
    h[0] = (__bf16)f[0]; h[1] = (__bf16)f[1];
    h[2] = (__bf16)f[2]; h[3] = (__bf16)f[3];
    *(v4bf*)(out + i) = h;
  }
}

// ---------------------------------------------------------------------------
// q-projection: C[m,n] = sum_k Q[m*QD+k] * W[n*QD+k]  (M = TQ*B rows).
// Block tile 64(M) x 256(N); 8 waves as 2(M) x 4(N); wave tile 32x64
// (8 WMMA per K-step from 2 A-fragments x 4 B-fragments).
// ---------------------------------------------------------------------------
__global__ __launch_bounds__(256) void qproj_kernel(
    const float* __restrict__ Q, const __bf16* __restrict__ Wb,
    __bf16* __restrict__ Qp) {
  const int lane = threadIdx.x & 31;
  const int wave = threadIdx.x >> 5;
  const int wrow = wave >> 2, wcol = wave & 3;
  const int hi = lane >> 4, lo = lane & 15;
  const int mW = blockIdx.x * 64 + wrow * 32;
  const int nW = blockIdx.y * 256 + wcol * 64;

  const float* aRow0 = Q + (size_t)(mW + lo) * QD;
  const float* aRow1 = Q + (size_t)(mW + 16 + lo) * QD;
  const __bf16* bCol[4];
#pragma unroll
  for (int j = 0; j < 4; ++j)
    bCol[j] = Wb + (size_t)(nW + j * 16 + lo) * QD + hi * 16;

  v8f c[2][4] = {};
  for (int k = 0; k < QD; k += 32) {
    __builtin_prefetch(bCol[0] + k + 256, 0, 1);   // global_prefetch_b8
    v16bf a[2];
    {
      v8f f0 = *(const v8f*)(aRow0 + k + hi * 16);
      v8f f1 = *(const v8f*)(aRow0 + k + 8 + hi * 16);
      v8f f2 = *(const v8f*)(aRow1 + k + hi * 16);
      v8f f3 = *(const v8f*)(aRow1 + k + 8 + hi * 16);
#pragma unroll
      for (int i = 0; i < 8; ++i) {
        a[0][i] = (__bf16)f0[i]; a[0][i + 8] = (__bf16)f1[i];
        a[1][i] = (__bf16)f2[i]; a[1][i + 8] = (__bf16)f3[i];
      }
    }
    v16bf bm[4];
#pragma unroll
    for (int j = 0; j < 4; ++j) bm[j] = *(const v16bf*)(bCol[j] + k);
#pragma unroll
    for (int i = 0; i < 2; ++i)
#pragma unroll
      for (int j = 0; j < 4; ++j) c[i][j] = wmma_bf16(a[i], bm[j], c[i][j]);
  }
#pragma unroll
  for (int i = 0; i < 2; ++i) {
    const int m0 = mW + i * 16 + hi * 8;
#pragma unroll
    for (int j = 0; j < 4; ++j) {
      const int n = nW + j * 16 + lo;
#pragma unroll
      for (int jj = 0; jj < 8; ++jj)
        Qp[(size_t)(m0 + jj) * KD + n] = (__bf16)c[i][j][jj];
    }
  }
}

// ---------------------------------------------------------------------------
// scores[t,b,s] = sum_k Qp[t,b,k] * K[s,b,k] (raw logits into d_out region).
// Same 64x256 block / 32x64 wave tiling, per batch b.
// ---------------------------------------------------------------------------
__global__ __launch_bounds__(256) void scores_kernel(
    const __bf16* __restrict__ Qp, const __bf16* __restrict__ Kb,
    float* __restrict__ S) {
  const int lane = threadIdx.x & 31;
  const int wave = threadIdx.x >> 5;
  const int wrow = wave >> 2, wcol = wave & 3;
  const int hi = lane >> 4, lo = lane & 15;
  const int b = blockIdx.z;
  const int tW = blockIdx.x * 64 + wrow * 32;
  const int sW = blockIdx.y * 256 + wcol * 64;

  const __bf16* aRow0 = Qp + ((size_t)(tW + lo) * BB + b) * KD;
  const __bf16* aRow1 = Qp + ((size_t)(tW + 16 + lo) * BB + b) * KD;
  const __bf16* bCol[4];
#pragma unroll
  for (int j = 0; j < 4; ++j)
    bCol[j] = Kb + ((size_t)(sW + j * 16 + lo) * BB + b) * KD + hi * 16;

  v8f c[2][4] = {};
  for (int k = 0; k < KD; k += 32) {
    __builtin_prefetch(bCol[0] + k + 256, 0, 1);
    v16bf a[2];
    {
      v8bf a00 = *(const v8bf*)(aRow0 + k + hi * 16);
      v8bf a01 = *(const v8bf*)(aRow0 + k + 8 + hi * 16);
      v8bf a10 = *(const v8bf*)(aRow1 + k + hi * 16);
      v8bf a11 = *(const v8bf*)(aRow1 + k + 8 + hi * 16);
#pragma unroll
      for (int i = 0; i < 8; ++i) {
        a[0][i] = a00[i]; a[0][i + 8] = a01[i];
        a[1][i] = a10[i]; a[1][i + 8] = a11[i];
      }
    }
    v16bf bm[4];
#pragma unroll
    for (int j = 0; j < 4; ++j) bm[j] = *(const v16bf*)(bCol[j] + k);
#pragma unroll
    for (int i = 0; i < 2; ++i)
#pragma unroll
      for (int j = 0; j < 4; ++j) c[i][j] = wmma_bf16(a[i], bm[j], c[i][j]);
  }
#pragma unroll
  for (int i = 0; i < 2; ++i) {
    const int t0 = tW + i * 16 + hi * 8;
#pragma unroll
    for (int j = 0; j < 4; ++j) {
      const int s = sW + j * 16 + lo;
#pragma unroll
      for (int jj = 0; jj < 8; ++jj)
        S[((size_t)(t0 + jj) * BB + b) * TK + s] = c[i][j][jj];
    }
  }
}

// ---------------------------------------------------------------------------
// Masked softmax over s (row length TK), in place, plus bf16 copy.
// ---------------------------------------------------------------------------
__global__ __launch_bounds__(256) void softmax_kernel(
    float* __restrict__ S, const unsigned char* __restrict__ mask,
    __bf16* __restrict__ Ab) {
  const int t = blockIdx.x, b = blockIdx.y;
  float*  row  = S  + ((size_t)t * BB + b) * TK;
  __bf16* arow = Ab + ((size_t)t * BB + b) * TK;
  const int tid = threadIdx.x;
  const int lane = tid & 31, wid = tid >> 5;

  float v[4];
#pragma unroll
  for (int i = 0; i < 4; ++i) {
    int s = tid + i * 256;
    float x = row[s];
    if (mask[(size_t)s * BB + b]) x = -__builtin_inff();
    v[i] = x;
  }
  float lm = fmaxf(fmaxf(v[0], v[1]), fmaxf(v[2], v[3]));
#pragma unroll
  for (int off = 16; off > 0; off >>= 1) lm = fmaxf(lm, __shfl_xor(lm, off));
  __shared__ float red[8];
  if (lane == 0) red[wid] = lm;
  __syncthreads();
  float m = red[0];
#pragma unroll
  for (int i = 1; i < 8; ++i) m = fmaxf(m, red[i]);

  float ls = 0.f;
#pragma unroll
  for (int i = 0; i < 4; ++i) { v[i] = __expf(v[i] - m); ls += v[i]; }
#pragma unroll
  for (int off = 16; off > 0; off >>= 1) ls += __shfl_xor(ls, off);
  __syncthreads();
  if (lane == 0) red[wid] = ls;
  __syncthreads();
  float sum = 0.f;
#pragma unroll
  for (int i = 0; i < 8; ++i) sum += red[i];
  float inv = 1.0f / sum;

#pragma unroll
  for (int i = 0; i < 4; ++i) {
    int s = tid + i * 256;
    float p = v[i] * inv;
    row[s]  = p;
    arow[s] = (__bf16)p;
  }
}

// ---------------------------------------------------------------------------
// x[t,b,v] = sum_s attn[t,b,s] * V[s,b,v].
// Block tile 64(t) x 256(v). Per K-step a 32(s) x 256(v) bf16 V tile is
// staged in LDS by the Tensor Data Mover (2D D#: data_size=2B, tile 256x32,
// dim0 stride B*VD), synchronized with s_wait_tensorcnt + barrier; B-matrix
// fragments are gathered (transposed) from LDS.
// ---------------------------------------------------------------------------
__global__ __launch_bounds__(256) void out_kernel(
    const __bf16* __restrict__ Ab, const __bf16* __restrict__ Vb,
    float* __restrict__ X) {
  __shared__ __bf16 vt[32 * 256];   // 16 KB V tile
  const int tid = threadIdx.x;
  const int lane = tid & 31, wave = tid >> 5;
  const int wrow = wave >> 2, wcol = wave & 3;
  const int hi = lane >> 4, lo = lane & 15;
  const int b = blockIdx.z;
  const int tW = blockIdx.x * 64 + wrow * 32;
  const int vBase = blockIdx.y * 256;

  const __bf16* aRow0 = Ab + ((size_t)(tW + lo) * BB + b) * TK;
  const __bf16* aRow1 = Ab + ((size_t)(tW + 16 + lo) * BB + b) * TK;

  v8f c[2][4] = {};
  for (int k = 0; k < TK; k += 32) {
#ifdef HAVE_TDM
    if (wave == 0) {
      unsigned long long gaddr = (unsigned long long)(uintptr_t)
          (Vb + ((size_t)k * BB + b) * VD + vBase);
      unsigned ldsOff = (unsigned)(uintptr_t)&vt[0];  // low 32 bits = LDS offset
      v4u g0;
      g0[0] = 1u;                                   // count=1 (valid user D#)
      g0[1] = ldsOff;                               // lds_addr
      g0[2] = (unsigned)(gaddr & 0xffffffffu);      // global_addr[31:0]
      g0[3] = (unsigned)((gaddr >> 32) & 0x1ffffffu) | (2u << 30); // [56:32]+type=2
      v8i g1;
      g1[0] = (1 << 16);                            // data_size = 2 bytes
      g1[1] = (int)(256u << 16);                    // tensor_dim0[15:0] = 256
      g1[2] = (int)(32u << 16);                     // dim0 hi=0 | tensor_dim1 lo=32
      g1[3] = (int)(256u << 16);                    // dim1 hi=0 | tile_dim0=256
      g1[4] = (int)(32u);                           // tile_dim1=32, tile_dim2=0
      g1[5] = (int)(BB * VD);                       // tensor_dim0_stride = 16384
      g1[6] = 0;                                    // stride hi | dim1_stride lo
      g1[7] = 0;
      v4i gz = {0, 0, 0, 0};
#if defined(__clang_major__) && __clang_major__ >= 23
      v8i gz8 = {0, 0, 0, 0, 0, 0, 0, 0};
      __builtin_amdgcn_tensor_load_to_lds(g0, g1, gz, gz, gz8, 0);
#else
      __builtin_amdgcn_tensor_load_to_lds(g0, g1, gz, gz, 0);
#endif
      __builtin_amdgcn_s_wait_tensorcnt(0);
    }
    __syncthreads();
#else
    // Cooperative fallback staging: 32(s) x 256(v), 8 bf16 per load.
#pragma unroll
    for (int r = 0; r < 4; ++r) {
      int chunk = tid + r * 256;     // 0..1023
      int si = chunk >> 5;           // 0..31
      int vi = (chunk & 31) * 8;     // 0..248
      *(v8bf*)(vt + si * 256 + vi) =
          *(const v8bf*)(Vb + ((size_t)(k + si) * BB + b) * VD + vBase + vi);
    }
    __syncthreads();
#endif
    v16bf a[2];
    {
      v8bf a00 = *(const v8bf*)(aRow0 + k + hi * 8);
      v8bf a01 = *(const v8bf*)(aRow0 + k + 16 + hi * 8);
      v8bf a10 = *(const v8bf*)(aRow1 + k + hi * 8);
      v8bf a11 = *(const v8bf*)(aRow1 + k + 16 + hi * 8);
#pragma unroll
      for (int i = 0; i < 8; ++i) {
        a[0][i] = a00[i]; a[0][i + 8] = a01[i];
        a[1][i] = a10[i]; a[1][i + 8] = a11[i];
      }
    }
    v16bf bm[4];
#pragma unroll
    for (int j = 0; j < 4; ++j) {
      const __bf16* col = vt + (hi * 8) * 256 + wcol * 64 + j * 16 + lo;
#pragma unroll
      for (int jj = 0; jj < 16; ++jj) bm[j][jj] = col[(jj + (jj >> 3) * 8) * 256];
    }
#pragma unroll
    for (int i = 0; i < 2; ++i)
#pragma unroll
      for (int j = 0; j < 4; ++j) c[i][j] = wmma_bf16(a[i], bm[j], c[i][j]);
    __syncthreads();
  }
#pragma unroll
  for (int i = 0; i < 2; ++i) {
    const int t0 = tW + i * 16 + hi * 8;
#pragma unroll
    for (int j = 0; j < 4; ++j) {
      const int v = vBase + wcol * 64 + j * 16 + lo;
#pragma unroll
      for (int jj = 0; jj < 8; ++jj)
        X[((size_t)(t0 + jj) * BB + b) * VD + v] = c[i][j][jj];
    }
  }
}

// ---------------------------------------------------------------------------
// Host launcher
// ---------------------------------------------------------------------------
extern "C" void kernel_launch(void* const* d_in, const int* in_sizes, int n_in,
                              void* d_out, int out_size, void* d_ws, size_t ws_size,
                              hipStream_t stream) {
  const float* dQ = (const float*)d_in[0];          // [TQ,B,QD]
  const float* dK = (const float*)d_in[1];          // [TK,B,KD]
  const float* dV = (const float*)d_in[2];          // [TK,B,VD]
  const float* dW = (const float*)d_in[3];          // [KD,QD]
  const unsigned char* dMask = (const unsigned char*)d_in[4];  // [TK,B] bool

  float* dX    = (float*)d_out;                     // [TQ,B,VD]
  float* dAttn = dX + (size_t)TQ * BB * VD;         // [TQ,B,TK]

  char* ws = (char*)d_ws;
  __bf16* Wb = (__bf16*)ws; ws += (size_t)KD * QD * 2;
  __bf16* Kb = (__bf16*)ws; ws += (size_t)TK * BB * KD * 2;
  __bf16* Vb = (__bf16*)ws; ws += (size_t)TK * BB * VD * 2;
  __bf16* Qp = (__bf16*)ws; ws += (size_t)TQ * BB * KD * 2;
  __bf16* Ab = (__bf16*)ws;

  cvt_f32_bf16<<<(KD * QD / 4 + 255) / 256, 256, 0, stream>>>(dW, Wb, KD * QD);
  cvt_f32_bf16<<<(TK * BB * KD / 4 + 255) / 256, 256, 0, stream>>>(dK, Kb, TK * BB * KD);
  cvt_f32_bf16<<<(TK * BB * VD / 4 + 255) / 256, 256, 0, stream>>>(dV, Vb, TK * BB * VD);

  qproj_kernel<<<dim3(TQ * BB / 64, KD / 256), 256, 0, stream>>>(dQ, Wb, Qp);
  scores_kernel<<<dim3(TQ / 64, TK / 256, BB), 256, 0, stream>>>(Qp, Kb, dAttn);
  softmax_kernel<<<dim3(TQ, BB), 256, 0, stream>>>(dAttn, dMask, Ab);
  out_kernel<<<dim3(TQ / 64, VD / 256, BB), 256, 0, stream>>>(Ab, Vb, dX);
}
